// GRU_9088150798389
// MI455X (gfx1250) — matrix operand order repaired and checked by reading, not compiled, not run on hardware
//
#include <hip/hip_runtime.h>
#include <hip/hip_bf16.h>

typedef __bf16 bf16_t;
typedef __attribute__((ext_vector_type(16))) __bf16 v16bf;
typedef __attribute__((ext_vector_type(8)))  float  v8f;

// ---------------------------------------------------------------------------
// WMMA fragment helpers, following CDNA5 ISA §7.12.2 VGPR layouts (wave32).
// A (16x32 bf16): lane L<16 holds M=L, K=(0..7,16..23); lane L+16 K=(8..15,24..31)
// B (32x16 bf16): lane L holds column N=L%16, K in the same order as A
//   -> store B panel k-major in LDS ; a lane reads at stride ld, a half-wave reads 16 contiguous bf16 per k.
// C (16x16 f32): lane L<16 col N=L rows M=v; lane L+16 rows M=v+8.
// ---------------------------------------------------------------------------
__device__ __forceinline__ v16bf frag_a(const bf16_t* s, int ld, int m0, int k0) {
  const int lane = threadIdx.x & 31;
  const bf16_t* p = s + (size_t)(m0 + (lane & 15)) * ld + k0 + ((lane >> 4) << 3);
  v16bf a;
#pragma unroll
  for (int i = 0; i < 8; ++i) a[i] = p[i];
#pragma unroll
  for (int i = 0; i < 8; ++i) a[8 + i] = p[16 + i];
  return a;
}

__device__ __forceinline__ v16bf frag_b(const bf16_t* w, int ld, int k0, int n0) {
  const int lane = threadIdx.x & 31;
  const bf16_t* p = w + (size_t)(k0 + ((lane >> 4) << 3)) * ld + n0 + (lane & 15);
  v16bf b;
#pragma unroll
  for (int i = 0; i < 16; ++i) b[i] = p[(size_t)(i + ((i >> 3) << 3)) * ld];
  return b;
}

__device__ __forceinline__ v8f wmma_bf16(v16bf a, v16bf b, v8f c) {
  // (neg_a, A, neg_b, B, c_mod, C, reuse_a, reuse_b)
  return __builtin_amdgcn_wmma_f32_16x16x32_bf16(false, a, false, b, (short)0, c,
                                                 false, false);
}

// Async global->LDS copy of 16 bytes per lane (CDNA5 GLOBAL_LOAD_ASYNC_TO_LDS,
// ASYNCcnt-tracked).  lds_off is the byte offset within the workgroup's LDS.
__device__ __forceinline__ void async_copy_b128(unsigned lds_off,
                                                const void* gaddr) {
  asm volatile("global_load_async_to_lds_b128 %0, %1, off"
               :: "v"(lds_off), "v"((unsigned long long)(size_t)gaddr)
               : "memory");
}
__device__ __forceinline__ void wait_async() {
  asm volatile("s_wait_asynccnt 0" ::: "memory");
}
__device__ __forceinline__ unsigned lds_offset_of(const void* p) {
  // Flat LDS aperture: addr[31:0] is the LDS byte offset (ISA 10.2).
  return (unsigned)(unsigned long long)(size_t)p;
}

__device__ __forceinline__ float fast_sigmoid(float x) {
  return 1.0f / (1.0f + __expf(-x));
}
__device__ __forceinline__ float fast_tanh(float x) {
  return 1.0f - 2.0f / (__expf(2.0f * x) + 1.0f);
}

// ---------------------------------------------------------------------------
// Fused GRU layer scan.  One workgroup owns a BT=32 batch tile for all T=120
// steps.  Weight panels live in LDS (bf16, k-major) for the whole kernel:
//   P_x [KX][3H] : [Wih_r | Wih_z | Wih_n]^T
//   P_h [H ][3H] : [Whh_r | Whh_z | Whh_n]^T
// Per step the fused GEMM writes g[BT][4H] = [r_pre, z_pre, xn, hn]:
//   chunks 0,1 accumulate over both panels, chunk 2 (xn) over P_x only,
//   chunk 3 (hn) over P_h cols [2H,3H) only.
// ---------------------------------------------------------------------------
template <int KIN, int H>
__global__ __launch_bounds__(256) void gru_scan_kernel(
    const float* __restrict__ Wih, const float* __restrict__ Whh,
    const float* __restrict__ bih, const float* __restrict__ bhh,
    const float* __restrict__ h0,
    const float* __restrict__ xf,   // layer-1 fp32 input [B,T,KIN] (or null)
    const bf16_t* __restrict__ xb,  // bf16 input [B,T,KIN]        (or null)
    bf16_t* __restrict__ out)       // [B,T,H] bf16
{
  constexpr int T   = 120;
  constexpr int BT  = 32;
  constexpr int KX  = (KIN + 31) & ~31;  // K padded to WMMA depth
  constexpr int PLD = 3 * H + 8;         // weight panel ld, bank-skewed
  constexpr int SLD = KX + H + 8;        // activation stage ld ([x | h] bf16)
  constexpr int GLD = 4 * H + 8;         // gate preact ld (f32)
  constexpr int HLD = H + 8;             // f32 hidden state ld

  constexpr size_t OPX = 0;
  constexpr size_t OPH = OPX + (size_t)KX * PLD * sizeof(bf16_t);
  constexpr size_t OST = OPH + (size_t)H  * PLD * sizeof(bf16_t);
  constexpr size_t OG  = OST + (size_t)BT * SLD * sizeof(bf16_t);
  constexpr size_t OH  = OG  + (size_t)BT * GLD * sizeof(float);
  constexpr size_t OB  = OH  + (size_t)BT * HLD * sizeof(float);

  extern __shared__ __attribute__((aligned(16))) char smem[];
  bf16_t* px   = (bf16_t*)(smem + OPX);
  bf16_t* ph   = (bf16_t*)(smem + OPH);
  bf16_t* st   = (bf16_t*)(smem + OST);
  float*  g    = (float*)(smem + OG);
  float*  hf   = (float*)(smem + OH);
  float*  bias = (float*)(smem + OB);

  const int tid = threadIdx.x;
  const int bg0 = blockIdx.x * BT;

  // ---- one-time staging: weights (f32 -> bf16, transposed k-major) ----
  for (int i = tid; i < KX * 3 * H; i += 256) {
    const int kx = i / (3 * H), n = i % (3 * H);
    const float v = (kx < KIN) ? Wih[(size_t)n * KIN + kx] : 0.0f;
    px[(size_t)kx * PLD + n] = (bf16_t)v;
  }
  for (int i = tid; i < H * 3 * H; i += 256) {
    const int kh = i / (3 * H), n = i % (3 * H);
    ph[(size_t)kh * PLD + n] = (bf16_t)Whh[(size_t)n * H + kh];
  }
  for (int c = tid; c < 4 * H; c += 256) {
    float v;
    if (c < 2 * H)      v = bih[c] + bhh[c];   // r,z: both biases fold
    else if (c < 3 * H) v = bih[c];            // xn: input bias only
    else                v = bhh[c - H];        // hn: hidden bias only
    bias[c] = v;
  }
  for (int i = tid; i < BT * H; i += 256) {
    const int b = i / H, j = i % H;
    const float v = h0[(size_t)(bg0 + b) * H + j];
    hf[b * HLD + j] = v;
    st[b * SLD + KX + j] = (bf16_t)v;
  }
  if (KX > KIN) {  // zero K padding (layer 1 only: 16 -> 32)
    for (int i = tid; i < BT * (KX - KIN); i += 256) {
      const int b = i / (KX - KIN), j = i % (KX - KIN);
      st[b * SLD + KIN + j] = (bf16_t)0.0f;
    }
  }
  __syncthreads();

  // Force wave id into an SGPR: all tile-loop control below becomes scalar
  // (s_cbranch), keeping EXEC all-1s around WMMA as the ISA requires.
  const int wave = __builtin_amdgcn_readfirstlane(tid >> 5);
  constexpr int MT = BT / 16;
  constexpr int NT = (4 * H) / 16;

  for (int t = 0; t < T; ++t) {
    // ---- stage x_t into [x | h] buffer ----
    if (xf) {  // layer 1: fp32 -> bf16 conversion (VALU copy)
      for (int i = tid; i < BT * KIN; i += 256) {
        const int b = i / KIN, j = i % KIN;
        st[b * SLD + j] = (bf16_t)xf[((size_t)(bg0 + b) * T + t) * KIN + j];
      }
      if (t + 1 < T && tid < BT)
        __builtin_prefetch(&xf[((size_t)(bg0 + tid) * T + t + 1) * KIN], 0, 0);
    } else {   // layers 2/3: bf16 -> bf16, async DMA straight into LDS
      for (int e = tid; e < (BT * KIN) / 8; e += 256) {
        const int b = e / (KIN / 8);
        const int j = (e % (KIN / 8)) * 8;  // 8 bf16 = 16 bytes per transfer
        async_copy_b128(lds_offset_of(&st[b * SLD + j]),
                        &xb[((size_t)(bg0 + b) * T + t) * KIN + j]);
      }
      if (t + 1 < T && tid < BT)
        __builtin_prefetch(&xb[((size_t)(bg0 + tid) * T + t + 1) * KIN], 0, 0);
      wait_async();
    }
    __syncthreads();

    // ---- fused gate GEMM: g[BT][4H] ----
    for (int tile = wave; tile < MT * NT; tile += 8) {
      const int mt = tile % MT, nt = tile / MT;
      const int n0 = nt * 16;
      const int chunk = n0 / H;  // 0=r 1=z 2=xn 3=hn  (scalar)
      v8f acc = (v8f)0.0f;
      if (chunk < 3) {  // x contribution (r, z, xn)
#pragma unroll
        for (int k = 0; k < KX; k += 32)
          acc = wmma_bf16(frag_a(st, SLD, mt * 16, k),
                          frag_b(px, PLD, k, n0), acc);
      }
      if (chunk != 2) {  // h contribution (r, z, hn)
        const int nph = (chunk == 3) ? (n0 - H) : n0;
#pragma unroll
        for (int k = 0; k < H; k += 32)
          acc = wmma_bf16(frag_a(st, SLD, mt * 16, KX + k),
                          frag_b(ph, PLD, k, nph), acc);
      }
      {  // C-fragment -> LDS (f32), per ISA C/D layout
        const int lane = tid & 31;
        const int col = n0 + (lane & 15);
        const int r0 = mt * 16 + ((lane >> 4) << 3);
#pragma unroll
        for (int v = 0; v < 8; ++v) g[(r0 + v) * GLD + col] = acc[v];
      }
    }
    __syncthreads();

    // ---- elementwise gates + state update ----
    for (int i = tid; i < BT * H; i += 256) {
      const int b = i / H, j = i % H;
      const float gr = g[b * GLD + j]         + bias[j];
      const float gz = g[b * GLD + H + j]     + bias[H + j];
      const float gx = g[b * GLD + 2 * H + j] + bias[2 * H + j];
      const float gh = g[b * GLD + 3 * H + j] + bias[3 * H + j];
      const float r = fast_sigmoid(gr);
      const float z = fast_sigmoid(gz);
      const float n = fast_tanh(gx + r * gh);
      const float hp = hf[b * HLD + j];
      const float hn = (1.0f - z) * n + z * hp;
      hf[b * HLD + j] = hn;
      st[b * SLD + KX + j] = (bf16_t)hn;
      out[((size_t)(bg0 + b) * T + t) * H + j] = (bf16_t)hn;
    }
    __syncthreads();
  }
}

template <int KIN, int H>
constexpr size_t scan_smem_bytes() {
  constexpr int BT = 32, KX = (KIN + 31) & ~31;
  constexpr int PLD = 3 * H + 8, SLD = KX + H + 8, GLD = 4 * H + 8, HLD = H + 8;
  return (size_t)KX * PLD * 2 + (size_t)H * PLD * 2 + (size_t)BT * SLD * 2 +
         (size_t)BT * GLD * 4 + (size_t)BT * HLD * 4 + (size_t)4 * H * 4;
}

// ---------------------------------------------------------------------------
// Tiled WMMA GEMM with bias: out[M,N(valid)] = A_bf16[M,K] @ W_f32[N,K]^T + b
// 64x64 block tile, BK=32, 8 waves x 2 accumulator tiles.
// A tiles are DMA'd global->LDS with async b128 copies (bf16 -> bf16).
// ---------------------------------------------------------------------------
template <bool F32OUT>
__global__ __launch_bounds__(256) void gemm_bias_kernel(
    int K, int Nvalid,
    const bf16_t* __restrict__ A, const float* __restrict__ W,
    const float* __restrict__ bias,
    float* __restrict__ outF, bf16_t* __restrict__ outB, int ldo)
{
  constexpr int BM = 64, BN = 64, BK = 32;
  __shared__ bf16_t As[BM][BK + 8];
  __shared__ bf16_t Bs[BK][BN + 8];
  const int tid = threadIdx.x;
  const int m0 = blockIdx.x * BM;
  const int n0 = blockIdx.y * BN;
  const int wave = __builtin_amdgcn_readfirstlane(tid >> 5);
  const int mt = wave & 3;        // 4 M-tiles
  const int ntA = wave >> 2;      // tiles ntA and ntA+2
  v8f acc0 = (v8f)0.0f, acc1 = (v8f)0.0f;

  for (int k0 = 0; k0 < K; k0 += BK) {
    {  // A tile: 64x32 bf16 = 256 async b128 transfers (1 per thread)
      const int m = tid >> 2;
      const int kk = (tid & 3) * 8;
      async_copy_b128(lds_offset_of(&As[m][kk]),
                      &A[(size_t)(m0 + m) * K + k0 + kk]);
    }
    for (int i = tid; i < BK * BN; i += 256) {  // W tile: f32 -> bf16 (VALU)
      const int kk = i & (BK - 1), n = i >> 5;
      const float v = (n0 + n < Nvalid) ? W[(size_t)(n0 + n) * K + k0 + kk] : 0.0f;
      Bs[kk][n] = (bf16_t)v;
    }
    wait_async();
    __syncthreads();
    const v16bf a  = frag_a(&As[0][0], BK + 8, mt * 16, 0);
    const v16bf b0 = frag_b(&Bs[0][0], BN + 8, 0, ntA * 16);
    const v16bf b1 = frag_b(&Bs[0][0], BN + 8, 0, (ntA + 2) * 16);
    acc0 = wmma_bf16(a, b0, acc0);
    acc1 = wmma_bf16(a, b1, acc1);
    __syncthreads();
  }

  const int lane = tid & 31;
  const int r0 = m0 + mt * 16 + ((lane >> 4) << 3);
#pragma unroll
  for (int half = 0; half < 2; ++half) {
    const v8f acc = half ? acc1 : acc0;
    const int cg = n0 + (half ? (ntA + 2) : ntA) * 16 + (lane & 15);
    if (cg < Nvalid) {
      const float bv = bias[cg];
#pragma unroll
      for (int v = 0; v < 8; ++v) {
        const float val = acc[v] + bv;
        if constexpr (F32OUT) outF[(size_t)(r0 + v) * ldo + cg] = val;
        else                  outB[(size_t)(r0 + v) * ldo + cg] = (bf16_t)val;
      }
    }
  }
}

// ---------------------------------------------------------------------------
extern "C" void kernel_launch(void* const* d_in, const int* in_sizes, int n_in,
                              void* d_out, int out_size, void* d_ws, size_t ws_size,
                              hipStream_t stream) {
  (void)in_sizes; (void)n_in; (void)out_size; (void)ws_size;
  const float* X    = (const float*)d_in[0];
  const float* h1   = (const float*)d_in[1];
  const float* h2   = (const float*)d_in[2];
  const float* h3   = (const float*)d_in[3];
  const float* Wih1 = (const float*)d_in[4];
  const float* Whh1 = (const float*)d_in[5];
  const float* bih1 = (const float*)d_in[6];
  const float* bhh1 = (const float*)d_in[7];
  const float* Wih2 = (const float*)d_in[8];
  const float* Whh2 = (const float*)d_in[9];
  const float* bih2 = (const float*)d_in[10];
  const float* bhh2 = (const float*)d_in[11];
  const float* Wih3 = (const float*)d_in[12];
  const float* Whh3 = (const float*)d_in[13];
  const float* bih3 = (const float*)d_in[14];
  const float* bhh3 = (const float*)d_in[15];
  const float* Wfc1 = (const float*)d_in[16];
  const float* bfc1 = (const float*)d_in[17];
  const float* Wfc2 = (const float*)d_in[18];
  const float* bfc2 = (const float*)d_in[19];

  // Workspace layout (bf16 intermediates; o3/f1 reuse o1's region once dead):
  //   o1 @ 0        : 60 MiB      o2 @ 64 MiB : 60 MiB
  //   o3 @ 0        : 30 MiB      f1 @ 36 MiB : 2 MiB
  char* ws = (char*)d_ws;
  bf16_t* o1 = (bf16_t*)(ws);
  bf16_t* o2 = (bf16_t*)(ws + (size_t)64 * 1024 * 1024);
  bf16_t* o3 = (bf16_t*)(ws);
  bf16_t* f1 = (bf16_t*)(ws + (size_t)36 * 1024 * 1024);
  float* out = (float*)d_out;

  constexpr int NBLK = 2048 / 32;  // 64 workgroups, 32-row batch tiles

  gru_scan_kernel<16, 128><<<NBLK, 256, scan_smem_bytes<16, 128>(), stream>>>(
      Wih1, Whh1, bih1, bhh1, h1, X, (const bf16_t*)nullptr, o1);
  gru_scan_kernel<128, 128><<<NBLK, 256, scan_smem_bytes<128, 128>(), stream>>>(
      Wih2, Whh2, bih2, bhh2, h2, (const float*)nullptr, o1, o2);
  gru_scan_kernel<128, 64><<<NBLK, 256, scan_smem_bytes<128, 64>(), stream>>>(
      Wih3, Whh3, bih3, bhh3, h3, (const float*)nullptr, o2, o3);

  // FC1: [2048,7680] x [512,7680]^T -> bf16 f1
  gemm_bias_kernel<false><<<dim3(2048 / 64, 512 / 64), 256, 0, stream>>>(
      7680, 512, o3, Wfc1, bfc1, nullptr, f1, 512);
  // FC2: [2048,512] x [50,512]^T -> f32 d_out (ld=50, masked to 50 cols)
  gemm_bias_kernel<true><<<dim3(2048 / 64, 1), 256, 0, stream>>>(
      512, 50, f1, Wfc2, bfc2, out, nullptr, 50);
}
